// GroupQueryAttention_81741817577827
// MI455X (gfx1250) — matrix operand order repaired and checked by reading, not compiled, not run on hardware
//
#include <hip/hip_runtime.h>
#include <hip/hip_bf16.h>

// ---------------- CDNA5 types / helpers ----------------
typedef __attribute__((ext_vector_type(8)))  float  v8f;
typedef __attribute__((ext_vector_type(8)))  __bf16 v8bf;
typedef __attribute__((ext_vector_type(16))) __bf16 v16bf;

#define SEQ   2048
#define DMOD  1024
#define HEADS 16
#define HD    64
#define HKV   4
#define DKV   256
#define NROWS 4096   // B*S

__device__ __forceinline__ v8f wmma_bf16(v16bf a, v16bf b, v8f c) {
    // D = A(16x32 bf16) * B(32x16 bf16) + C(16x16 f32)
    return __builtin_amdgcn_wmma_f32_16x16x32_bf16(
        /*neg_a=*/false, a, /*neg_b=*/false, b,
        /*c_mod=*/(short)0, c, /*reuse_a=*/false, /*reuse_b=*/false);
}

__device__ __forceinline__ v16bf cat8(v8bf lo, v8bf hi) {
    v16bf r;
#pragma unroll
    for (int i = 0; i < 8; ++i) { r[i] = lo[i]; r[i + 8] = hi[i]; }
    return r;
}

__device__ __forceinline__ void wait_ds0() {
#if __has_builtin(__builtin_amdgcn_s_wait_dscnt)
    __builtin_amdgcn_s_wait_dscnt(0);
#else
    asm volatile("s_wait_dscnt 0" ::: "memory");
#endif
}

// ---------------- elementwise conversion kernels ----------------
__global__ void gqa_f32_to_bf16(const float* __restrict__ in,
                                __bf16* __restrict__ out, int n) {
    int i = blockIdx.x * 256 + threadIdx.x;
    if (i < n) out[i] = (__bf16)in[i];
}

// W is (K x Nout) row-major fp32 -> Wt is (Nout x K) row-major bf16
__global__ void gqa_transpose_bf16(const float* __restrict__ W,
                                   __bf16* __restrict__ Wt, int K, int Nout) {
    int i = blockIdx.x * 256 + threadIdx.x;
    if (i >= K * Nout) return;
    int n = i / K, k = i % K;
    Wt[i] = (__bf16)W[(size_t)k * Nout + n];
}

// RoPE over full D for q: pair index p in [0,512)
__global__ void gqa_rope_q(const __bf16* __restrict__ in,
                           __bf16* __restrict__ out) {
    int i = blockIdx.x * 256 + threadIdx.x;      // over NROWS*512
    if (i >= NROWS * 512) return;
    int n = i >> 9, p = i & 511;
    int s = n & (SEQ - 1);
    float inv = __expf(-(float)p * (9.210340371976184f / 512.0f)); // theta^(-2p/D)
    float ang = (float)s * inv;
    float c = __cosf(ang), sn = __sinf(ang);
    size_t base = (size_t)n * DMOD + 2 * p;
    float x0 = (float)in[base], x1 = (float)in[base + 1];
    out[base]     = (__bf16)(x0 * c - x1 * sn);
    out[base + 1] = (__bf16)(x0 * sn + x1 * c);
}

// RoPE per kv-head for k: head-local pair j in [0,32), same freqs inv[j]
__global__ void gqa_rope_k(const __bf16* __restrict__ in,
                           __bf16* __restrict__ out) {
    int i = blockIdx.x * 256 + threadIdx.x;      // over NROWS*128
    if (i >= NROWS * 128) return;
    int n = i >> 7, rem = i & 127;
    int kvh = rem >> 5, j = rem & 31;
    int s = n & (SEQ - 1);
    float inv = __expf(-(float)j * (9.210340371976184f / 512.0f));
    float ang = (float)s * inv;
    float c = __cosf(ang), sn = __sinf(ang);
    size_t base = (size_t)n * DKV + kvh * HD + 2 * j;
    float x0 = (float)in[base], x1 = (float)in[base + 1];
    out[base]     = (__bf16)(x0 * c - x1 * sn);
    out[base + 1] = (__bf16)(x0 * sn + x1 * c);
}

// v (n=b*S+s, 256) -> vt[((b*HKV+kvh)*HD + d) * SEQ + s]  (K-contiguous for WMMA B)
__global__ void gqa_vtrans(const __bf16* __restrict__ vlin,
                           __bf16* __restrict__ vt) {
    int i = blockIdx.x * 256 + threadIdx.x;      // over NROWS*256
    if (i >= NROWS * DKV) return;
    int n = i >> 8, d256 = i & 255;
    int b = n >> 11, s = n & (SEQ - 1);
    int kvh = d256 >> 6, d = d256 & 63;
    vt[((size_t)((b * HKV + kvh) * HD + d)) * SEQ + s] = vlin[i];
}

// ------- bf16 WMMA GEMM: C(MxN) = A(MxK) * Bt(NxK)^T, 16x64 strip per wave -------
template <typename OutT>
__global__ __launch_bounds__(128)
void gqa_gemm_bf16(const __bf16* __restrict__ A,
                   const __bf16* __restrict__ Bt,
                   OutT* __restrict__ C, int M, int N, int K) {
    const int lane = threadIdx.x & 31;
    const int wave = blockIdx.x * (blockDim.x >> 5) + (threadIdx.x >> 5);
    const int ntn = N >> 6;                       // 64-wide N strips
    if (wave >= (M >> 4) * ntn) return;
    const int tm = wave / ntn, tn = wave % ntn;
    const int half = lane >> 4, l16 = lane & 15;

    // A fragment: lane-half holds K {0..7, 16..23} (+half*8)
    const __bf16* aRow = A + (size_t)(tm * 16 + l16) * K + half * 8;
    // B fragments: lane-half holds K {0..7, 16..23} (+half*8)
    const __bf16* bRow[4];
#pragma unroll
    for (int s = 0; s < 4; ++s)
        bRow[s] = Bt + (size_t)(tn * 64 + s * 16 + l16) * K + half * 8;

    v8f vzero = {};
    v8f acc[4];
#pragma unroll
    for (int s = 0; s < 4; ++s) acc[s] = vzero;

    for (int k0 = 0; k0 < K; k0 += 32) {
        v8bf aLo = *reinterpret_cast<const v8bf*>(aRow + k0);
        v8bf aHi = *reinterpret_cast<const v8bf*>(aRow + k0 + 16);
        v16bf afrag = cat8(aLo, aHi);
#pragma unroll
        for (int s = 0; s < 4; ++s) {             // 4 independent WMMAs share A
            v8bf bLo = *reinterpret_cast<const v8bf*>(bRow[s] + k0);
            v8bf bHi = *reinterpret_cast<const v8bf*>(bRow[s] + k0 + 16);
            acc[s] = wmma_bf16(afrag, cat8(bLo, bHi), acc[s]);
        }
    }
#pragma unroll
    for (int s = 0; s < 4; ++s) {
#pragma unroll
        for (int r = 0; r < 8; ++r) {
            int m = tm * 16 + r + 8 * half;
            C[(size_t)m * N + (tn * 64 + s * 16 + l16)] = (OutT)acc[s][r];
        }
    }
}

// ------- flash attention: one (b,h,16-query) tile per wave, 64-key chunks -------
__global__ __launch_bounds__(128)
void gqa_attn(const __bf16* __restrict__ q,   // (NROWS, 1024) roped
              const __bf16* __restrict__ k,   // (NROWS, 256)  roped
              const __bf16* __restrict__ vt,  // (B*HKV*HD, SEQ)
              __bf16* __restrict__ o) {       // (NROWS, 1024)
    const int lane = threadIdx.x & 31;
    const int wib  = threadIdx.x >> 5;
    const int wave = blockIdx.x * 4 + wib;
    if (wave >= 2 * HEADS * (SEQ / 16)) return;
    const int qt = wave & 127;            // S/16 tiles
    const int h  = (wave >> 7) & 15;
    const int b  = wave >> 11;
    const int half = lane >> 4, l16 = lane & 15;
    const int kvh = h >> 2;               // G = 4

    __shared__ __bf16 pshm[4 * 16 * 64];  // per-wave 16x64 p tile
    __bf16* pbuf = pshm + wib * 1024;

    const int n0 = b * SEQ + qt * 16;

    // q A-fragments for hd 0..31 and 32..63 (loaded once)
    const __bf16* qrow = q + (size_t)(n0 + l16) * DMOD + h * HD + half * 8;
    v16bf qa0 = cat8(*(const v8bf*)(qrow),      *(const v8bf*)(qrow + 16));
    v16bf qa1 = cat8(*(const v8bf*)(qrow + 32), *(const v8bf*)(qrow + 48));

    // ones matrix for WMMA row-sum (p @ 1 -> row sums replicated per lane group)
    v16bf onesB;
#pragma unroll
    for (int i = 0; i < 16; ++i) onesB[i] = (__bf16)1.0f;

    float row_max[8], row_sum[8];
    v8f acc[4];
    v8f vzero = {};
#pragma unroll
    for (int r = 0; r < 8; ++r) { row_max[r] = -__builtin_inff(); row_sum[r] = 0.0f; }
#pragma unroll
    for (int s4 = 0; s4 < 4; ++s4) acc[s4] = vzero;

    const float scale = 0.03125f;  // 1/sqrt(D), D = 1024

    for (int key0 = 0; key0 < SEQ; key0 += 64) {
        if (key0 + 64 < SEQ)  // warm L2/WGP$ for the next key chunk
            __builtin_prefetch(k + (size_t)(b * SEQ + key0 + 64 + lane) * DKV, 0, 1);

        // ---- scores: four 16-key tiles, K over hd=64 (2 WMMAs each) ----
        v8f sc[4];
#pragma unroll
        for (int t = 0; t < 4; ++t) {
            const __bf16* krow = k + (size_t)(b * SEQ + key0 + t * 16 + l16) * DKV
                                   + kvh * HD + half * 8;
            v16bf kb0 = cat8(*(const v8bf*)(krow),      *(const v8bf*)(krow + 16));
            v16bf kb1 = cat8(*(const v8bf*)(krow + 32), *(const v8bf*)(krow + 48));
            v8f s_ = vzero;
            s_ = wmma_bf16(qa0, kb0, s_);
            s_ = wmma_bf16(qa1, kb1, s_);
            sc[t] = s_;
        }

        // ---- online softmax: max via shuffle butterfly, exp -> LDS p tile ----
        float fvec[8];
#pragma unroll
        for (int r = 0; r < 8; ++r) {
            float a0 = sc[0][r] * scale, a1 = sc[1][r] * scale;
            float a2 = sc[2][r] * scale, a3 = sc[3][r] * scale;
            float mx = fmaxf(fmaxf(a0, a1), fmaxf(a2, a3));
#pragma unroll
            for (int m = 1; m < 16; m <<= 1) mx = fmaxf(mx, __shfl_xor(mx, m, 32));
            float nm = fmaxf(row_max[r], mx);
            float f  = __expf(row_max[r] - nm);
            row_max[r] = nm;
            fvec[r] = f;
            int mrow = r + 8 * half;
            pbuf[mrow * 64 + l16]      = (__bf16)__expf(a0 - nm);
            pbuf[mrow * 64 + 16 + l16] = (__bf16)__expf(a1 - nm);
            pbuf[mrow * 64 + 32 + l16] = (__bf16)__expf(a2 - nm);
            pbuf[mrow * 64 + 48 + l16] = (__bf16)__expf(a3 - nm);
#pragma unroll
            for (int s4 = 0; s4 < 4; ++s4) acc[s4][r] *= f;
        }
        wait_ds0();  // DS in-order within wave; make VGPR deps explicit

        // ---- p A-fragments (keys 0..31 and 32..63 of the chunk) ----
        const __bf16* prow = pbuf + l16 * 64 + half * 8;
        v16bf pa0 = cat8(*(const v8bf*)(prow),      *(const v8bf*)(prow + 16));
        v16bf pa1 = cat8(*(const v8bf*)(prow + 32), *(const v8bf*)(prow + 48));

        // ---- exact row sums via WMMA against ones (no shuffle reduction) ----
        v8f ssum = vzero;
        ssum = wmma_bf16(pa0, onesB, ssum);
        ssum = wmma_bf16(pa1, onesB, ssum);
#pragma unroll
        for (int r = 0; r < 8; ++r) row_sum[r] = row_sum[r] * fvec[r] + ssum[r];

        // ---- o += p @ v : 4 hd slices x 2 key sub-chunks (K=32 each) ----
#pragma unroll
        for (int s4 = 0; s4 < 4; ++s4) {
            const __bf16* vbase = vt + (size_t)((b * HKV + kvh) * HD + s4 * 16 + l16) * SEQ
                                     + key0 + half * 8;
            v16bf vb0 = cat8(*(const v8bf*)(vbase),      *(const v8bf*)(vbase + 16));
            v16bf vb1 = cat8(*(const v8bf*)(vbase + 32), *(const v8bf*)(vbase + 48));
            acc[s4] = wmma_bf16(pa0, vb0, acc[s4]);
            acc[s4] = wmma_bf16(pa1, vb1, acc[s4]);
        }
    }

    // ---- normalize and write bf16 output ----
#pragma unroll
    for (int r = 0; r < 8; ++r) {
        float inv = 1.0f / row_sum[r];
        int m = r + 8 * half;
#pragma unroll
        for (int s4 = 0; s4 < 4; ++s4) {
            o[(size_t)(n0 + m) * DMOD + h * HD + s4 * 16 + l16] =
                (__bf16)(acc[s4][r] * inv);
        }
    }
}

// ---------------- host launch ----------------
extern "C" void kernel_launch(void* const* d_in, const int* in_sizes, int n_in,
                              void* d_out, int out_size, void* d_ws, size_t ws_size,
                              hipStream_t stream) {
    (void)in_sizes; (void)n_in; (void)out_size; (void)ws_size;
    const float* x  = (const float*)d_in[0];
    const float* Wq = (const float*)d_in[1];
    const float* Wk = (const float*)d_in[2];
    const float* Wv = (const float*)d_in[3];
    const float* Wo = (const float*)d_in[4];

    char* ws = (char*)d_ws;
    size_t off = 0;
    auto carve = [&](size_t bytes) -> __bf16* {
        __bf16* p = (__bf16*)(ws + off);
        off += (bytes + 255) & ~(size_t)255;
        return p;
    };
    __bf16* xb    = carve((size_t)NROWS * DMOD * 2);  // 8 MiB
    __bf16* wqt   = carve((size_t)DMOD * DMOD * 2);   // 2 MiB
    __bf16* wkt   = carve((size_t)DKV  * DMOD * 2);
    __bf16* wvt   = carve((size_t)DKV  * DMOD * 2);
    __bf16* wot   = carve((size_t)DMOD * DMOD * 2);
    __bf16* qlin  = carve((size_t)NROWS * DMOD * 2);
    __bf16* qrope = carve((size_t)NROWS * DMOD * 2);
    __bf16* klin  = carve((size_t)NROWS * DKV * 2);
    __bf16* krope = carve((size_t)NROWS * DKV * 2);
    __bf16* vlin  = carve((size_t)NROWS * DKV * 2);
    __bf16* vtb   = carve((size_t)2 * HKV * HD * SEQ * 2);
    __bf16* obf   = carve((size_t)NROWS * DMOD * 2);  // total ~45 MiB

    // conversions
    {   int n = NROWS * DMOD;
        gqa_f32_to_bf16<<<(n + 255) / 256, 256, 0, stream>>>(x, xb, n); }
    gqa_transpose_bf16<<<(DMOD * DMOD + 255) / 256, 256, 0, stream>>>(Wq, wqt, DMOD, DMOD);
    gqa_transpose_bf16<<<(DMOD * DKV  + 255) / 256, 256, 0, stream>>>(Wk, wkt, DMOD, DKV);
    gqa_transpose_bf16<<<(DMOD * DKV  + 255) / 256, 256, 0, stream>>>(Wv, wvt, DMOD, DKV);
    gqa_transpose_bf16<<<(DMOD * DMOD + 255) / 256, 256, 0, stream>>>(Wo, wot, DMOD, DMOD);

    // projections (bf16 in, bf16 out, f32 accum); 16x64 strip per wave
    {   int waves = (NROWS / 16) * (DMOD / 64);
        gqa_gemm_bf16<__bf16><<<waves / 4, 128, 0, stream>>>(xb, wqt, qlin, NROWS, DMOD, DMOD); }
    {   int waves = (NROWS / 16) * (DKV / 64);
        gqa_gemm_bf16<__bf16><<<waves / 4, 128, 0, stream>>>(xb, wkt, klin, NROWS, DKV, DMOD); }
    {   int waves = (NROWS / 16) * (DKV / 64);
        gqa_gemm_bf16<__bf16><<<waves / 4, 128, 0, stream>>>(xb, wvt, vlin, NROWS, DKV, DMOD); }

    // RoPE + v transpose
    gqa_rope_q<<<(NROWS * 512 + 255) / 256, 256, 0, stream>>>(qlin, qrope);
    gqa_rope_k<<<(NROWS * 128 + 255) / 256, 256, 0, stream>>>(klin, krope);
    gqa_vtrans<<<(NROWS * DKV + 255) / 256, 256, 0, stream>>>(vlin, vtb);

    // attention: 2*16*128 = 4096 waves, 4 waves/block
    gqa_attn<<<1024, 128, 0, stream>>>(qrope, krope, vtb, obf);

    // output projection -> fp32 d_out
    {   int waves = (NROWS / 16) * (DMOD / 64);
        gqa_gemm_bf16<float><<<waves / 4, 128, 0, stream>>>(obf, wot, (float*)d_out,
                                                            NROWS, DMOD, DMOD); }
}
